// Moondream3TextMoE_54924041781498
// MI455X (gfx1250) — compile-verified, matrix operand order checked by reading, not run on hardware
//
#include <hip/hip_runtime.h>
#include <hip/hip_bf16.h>
#include <math.h>

#define HIDDEN 2048
#define INNER  1024
#define NEXP   64
#define TOPK   8
#define NTOK   512

typedef __attribute__((ext_vector_type(16))) __bf16 v16bf;
typedef __attribute__((ext_vector_type(8)))  float  v8f;

__device__ __forceinline__ __bf16 f2bf(float f) { return (__bf16)f; }

__device__ __forceinline__ v8f wmma_bf16(v16bf a, v16bf b, v8f c) {
  // (neg_a, A, neg_b, B, c_mod, C, reuse_a, reuse_b)
  return __builtin_amdgcn_wmma_f32_16x16x32_bf16(false, a, false, b, (short)0, c,
                                                 false, false);
}

__device__ __forceinline__ void async_wait0() {
#if __has_builtin(__builtin_amdgcn_s_wait_asynccnt)
  __builtin_amdgcn_s_wait_asynccnt(0);
#else
  asm volatile("s_wait_asynccnt 0" ::: "memory");
#endif
}

__device__ __forceinline__ void async_wait4() {
#if __has_builtin(__builtin_amdgcn_s_wait_asynccnt)
  __builtin_amdgcn_s_wait_asynccnt(4);
#else
  asm volatile("s_wait_asynccnt 4" ::: "memory");
#endif
}

// Per-lane async DMA of 16 bytes global -> LDS (ASYNCcnt-tracked).
__device__ __forceinline__ void async_copy_b128(unsigned lds_off, const void* gptr) {
  asm volatile("global_load_async_to_lds_b128 %0, %1, off"
               :: "v"(lds_off), "v"((unsigned long long)(uintptr_t)gptr)
               : "memory");
}

// Issue 4 x 32-lane b128 async DMAs: gather 16 token rows x 32 fp32 into LDS buf.
__device__ __forceinline__ void stage_chunk_async(const float* __restrict__ x,
                                                  const int* sTok,
                                                  float (*buf)[32], int k0, int lane) {
#pragma unroll
  for (int i = 0; i < 4; i++) {
    int r = i * 4 + (lane >> 3);
    int c = (lane & 7) << 2;
    const float* gp = x + (size_t)sTok[r] * HIDDEN + k0 + c;
    async_copy_b128((unsigned)(uintptr_t)&buf[r][c], gp);
  }
}

// Load a 16(N) x 32(K) B fragment from row-major W[n][k] (ldk = row stride),
// converting fp32 -> bf16 in-flight. Lane l holds output row n0 + (l&15),
// 16 contiguous K values starting at k0 + (l>>4)*16.
__device__ __forceinline__ v16bf load_b_frag(const float* __restrict__ W, int n0,
                                             int k0, int lane, int ldk) {
  const float* p = W + (size_t)(n0 + (lane & 15)) * (size_t)ldk + k0 + ((lane >> 4) << 4);
  __builtin_prefetch(p + 32, 0, 1);  // next K-chunk of this row -> global_prefetch_b8
  float4 f0 = ((const float4*)p)[0];
  float4 f1 = ((const float4*)p)[1];
  float4 f2 = ((const float4*)p)[2];
  float4 f3 = ((const float4*)p)[3];
  v16bf r;
  r[0]  = f2bf(f0.x); r[1]  = f2bf(f0.y); r[2]  = f2bf(f0.z); r[3]  = f2bf(f0.w);
  r[4]  = f2bf(f1.x); r[5]  = f2bf(f1.y); r[6]  = f2bf(f1.z); r[7]  = f2bf(f1.w);
  r[8]  = f2bf(f2.x); r[9]  = f2bf(f2.y); r[10] = f2bf(f2.z); r[11] = f2bf(f2.w);
  r[12] = f2bf(f3.x); r[13] = f2bf(f3.y); r[14] = f2bf(f3.z); r[15] = f2bf(f3.w);
  return r;
}

// ---------------------------------------------------------------------------
// Kernel 1: zero output accumulator and expert counters
// ---------------------------------------------------------------------------
__global__ void moe_init_kernel(float* __restrict__ out, int* __restrict__ counts,
                                int total) {
  int i = blockIdx.x * blockDim.x + threadIdx.x;
  if (i < total) out[i] = 0.0f;
  if (i < NEXP) counts[i] = 0;
}

// ---------------------------------------------------------------------------
// Kernel 2: gating — one wave per token (wave32; each lane owns 2 experts)
// softmax -> top-8 -> normalized weights -> scatter into per-expert lists
// ---------------------------------------------------------------------------
__global__ __launch_bounds__(256) void moe_gate_kernel(
    const float* __restrict__ x, const float* __restrict__ gw,
    const float* __restrict__ gb, int* __restrict__ counts,
    int* __restrict__ elist, float* __restrict__ ewt) {
  const int lane = threadIdx.x & 31;
  const int wave = threadIdx.x >> 5;
  const int t = blockIdx.x * 8 + wave;

  __shared__ float probs[8][NEXP];

  const int e0 = 2 * lane, e1 = 2 * lane + 1;
  float acc0 = gb[e0], acc1 = gb[e1];
  const float* xr = x + (size_t)t * HIDDEN;
  const float* g0 = gw + (size_t)e0 * HIDDEN;
  const float* g1 = gw + (size_t)e1 * HIDDEN;
  for (int i = 0; i < HIDDEN; i += 4) {
    float4 xv = *(const float4*)(xr + i);
    float4 a  = *(const float4*)(g0 + i);
    float4 b  = *(const float4*)(g1 + i);
    acc0 += xv.x * a.x + xv.y * a.y + xv.z * a.z + xv.w * a.w;
    acc1 += xv.x * b.x + xv.y * b.y + xv.z * b.z + xv.w * b.w;
  }
  // wave32 softmax over the 64 logits (2 per lane)
  float m = fmaxf(acc0, acc1);
  for (int o = 16; o > 0; o >>= 1) m = fmaxf(m, __shfl_xor(m, o, 32));
  float p0 = __expf(acc0 - m), p1 = __expf(acc1 - m);
  float s = p0 + p1;
  for (int o = 16; o > 0; o >>= 1) s += __shfl_xor(s, o, 32);
  float inv = 1.0f / s;
  probs[wave][e0] = p0 * inv;
  probs[wave][e1] = p1 * inv;
  __syncthreads();

  if (lane == 0) {
    int   idx[TOPK];
    float wv[TOPK];
    float wsum = 0.0f;
    for (int k = 0; k < TOPK; k++) {
      float best = -1.0f; int bi = 0;
      for (int e = 0; e < NEXP; e++) {
        float v = probs[wave][e];
        if (v > best) { best = v; bi = e; }
      }
      idx[k] = bi; wv[k] = best; wsum += best;
      probs[wave][bi] = -2.0f;
    }
    float winv = 1.0f / wsum;
    for (int k = 0; k < TOPK; k++) {
      int pos = atomicAdd(&counts[idx[k]], 1);
      elist[idx[k] * NTOK + pos] = t;
      ewt  [idx[k] * NTOK + pos] = wv[k] * winv;
    }
  }
}

// ---------------------------------------------------------------------------
// Kernel 3: per-expert SwiGLU MLP on 16-token tiles, bf16 WMMA
// grid = (32 tiles, 64 experts), 256 threads = 8 waves
// X chunks double-buffered into LDS via GLOBAL_LOAD_ASYNC_TO_LDS_B128.
// ---------------------------------------------------------------------------
__global__ __launch_bounds__(256, 1) void moe_expert_kernel(
    const float* __restrict__ x, const float* __restrict__ w1,
    const float* __restrict__ w2, const int* __restrict__ counts,
    const int* __restrict__ elist, const float* __restrict__ ewt,
    float* __restrict__ out) {
  const int e    = blockIdx.y;
  const int tile = blockIdx.x;
  const int cnt  = counts[e];
  if (tile * 16 >= cnt) return;
  const int nt = min(16, cnt - tile * 16);

  const int tid  = threadIdx.x;
  const int lane = tid & 31;
  const int wave = tid >> 5;

  __shared__ __align__(64) __bf16 sAct[16][INNER];   // 32 KB activation tile
  __shared__ __align__(64) float  sXf[2][16][32];    // 2 x 2 KB ping-pong A-chunks
  __shared__ int   sTok[16];
  __shared__ float sWt[16];

  if (tid < 16) {
    if (tid < nt) {
      sTok[tid] = elist[e * NTOK + tile * 16 + tid];
      sWt [tid] = ewt [e * NTOK + tile * 16 + tid];
    } else {
      sTok[tid] = elist[e * NTOK + tile * 16];  // valid address, weight 0
      sWt [tid] = 0.0f;
    }
  }
  __syncthreads();

  // ---------------- GEMM1: H = Xtile(16x2048) @ W1^T, fused SwiGLU --------
  // Wave w owns gate columns [w*128, w*128+128) and up columns
  // [1024 + w*128, ...), i.e. 8 gate tiles + 8 up tiles, so the activation
  // is computed entirely in accumulator registers.
  v8f accG[8] = {}, accU[8] = {};
  const float* W1e = w1 + (size_t)e * (2 * INNER) * HIDDEN;

  if (wave == 0) stage_chunk_async(x, sTok, sXf[0], 0, lane);  // prologue DMA

  for (int k0 = 0; k0 < HIDDEN; k0 += 32) {
    const int cur = (k0 >> 5) & 1;
    __syncthreads();  // all reads of sXf[cur^1] from the previous iteration done
    if (wave == 0) {
      if (k0 + 32 < HIDDEN) {
        stage_chunk_async(x, sTok, sXf[cur ^ 1], k0 + 32, lane);  // overlap next DMA
        async_wait4();  // only current chunk's 4 DMAs must have landed
      } else {
        async_wait0();
      }
    }
    __syncthreads();  // sXf[cur] ready for all waves
    // build bf16 A fragment from fp32 LDS chunk
    const float* ap = &sXf[cur][lane & 15][(lane >> 4) << 4];
    float4 a0 = ((const float4*)ap)[0];
    float4 a1 = ((const float4*)ap)[1];
    float4 a2 = ((const float4*)ap)[2];
    float4 a3 = ((const float4*)ap)[3];
    v16bf a;
    a[0]  = f2bf(a0.x); a[1]  = f2bf(a0.y); a[2]  = f2bf(a0.z); a[3]  = f2bf(a0.w);
    a[4]  = f2bf(a1.x); a[5]  = f2bf(a1.y); a[6]  = f2bf(a1.z); a[7]  = f2bf(a1.w);
    a[8]  = f2bf(a2.x); a[9]  = f2bf(a2.y); a[10] = f2bf(a2.z); a[11] = f2bf(a2.w);
    a[12] = f2bf(a3.x); a[13] = f2bf(a3.y); a[14] = f2bf(a3.z); a[15] = f2bf(a3.w);
#pragma unroll
    for (int j = 0; j < 8; j++) {
      int nG = wave * 128 + j * 16;
      int nU = INNER + nG;
      v16bf bG = load_b_frag(W1e, nG, k0, lane, HIDDEN);
      accG[j] = wmma_bf16(a, bG, accG[j]);
      v16bf bU = load_b_frag(W1e, nU, k0, lane, HIDDEN);
      accU[j] = wmma_bf16(a, bU, accU[j]);
    }
  }

  // SwiGLU (tanh-gelu gate * up) in registers, write bf16 act tile to LDS.
  __syncthreads();
#pragma unroll
  for (int j = 0; j < 8; j++) {
    int nbase = wave * 128 + j * 16;
#pragma unroll
    for (int r = 0; r < 8; r++) {
      float g = accG[j][r], u = accU[j][r];
      float t = 0.7978845608f * (g + 0.044715f * g * g * g);
      float act = 0.5f * g * (1.0f + tanhf(t)) * u;
      int mrow = r + ((lane >> 4) << 3);  // C-layout: M = r + 8*(lane/16)
      int col  = nbase + (lane & 15);     //            N = lane%16
      sAct[mrow][col] = f2bf(act);
    }
  }
  __syncthreads();

  // ---------------- GEMM2: Y = act(16x1024) @ W2^T (16x2048) --------------
  v8f accY[16] = {};
  const float* W2e = w2 + (size_t)e * HIDDEN * INNER;
  for (int k0 = 0; k0 < INNER; k0 += 32) {
    v16bf a = *(const v16bf*)(&sAct[lane & 15][k0 + ((lane >> 4) << 4)]);
#pragma unroll
    for (int j = 0; j < 16; j++) {
      int n = wave * 256 + j * 16;
      v16bf b = load_b_frag(W2e, n, k0, lane, INNER);
      accY[j] = wmma_bf16(a, b, accY[j]);
    }
  }

  // ---------------- combine-weighted scatter-add to output ----------------
#pragma unroll
  for (int j = 0; j < 16; j++) {
    int n = wave * 256 + j * 16 + (lane & 15);
#pragma unroll
    for (int r = 0; r < 8; r++) {
      int mrow = r + ((lane >> 4) << 3);
      if (mrow < nt) {
        atomicAdd(&out[(size_t)sTok[mrow] * HIDDEN + n], sWt[mrow] * accY[j][r]);
      }
    }
  }
}

// ---------------------------------------------------------------------------
extern "C" void kernel_launch(void* const* d_in, const int* in_sizes, int n_in,
                              void* d_out, int out_size, void* d_ws, size_t ws_size,
                              hipStream_t stream) {
  const float* x  = (const float*)d_in[0];
  const float* gw = (const float*)d_in[1];
  const float* gb = (const float*)d_in[2];
  const float* w1 = (const float*)d_in[3];
  const float* w2 = (const float*)d_in[4];
  float* out = (float*)d_out;

  // workspace layout
  int*   counts = (int*)d_ws;                     // 64
  int*   elist  = counts + NEXP;                  // 64*512 token ids
  float* ewt    = (float*)(elist + NEXP * NTOK);  // 64*512 combine weights

  int total = NTOK * HIDDEN;
  moe_init_kernel<<<(total + 255) / 256, 256, 0, stream>>>(out, counts, total);
  moe_gate_kernel<<<NTOK / 8, 256, 0, stream>>>(x, gw, gb, counts, elist, ewt);
  dim3 grid(NTOK / 16, NEXP);
  moe_expert_kernel<<<grid, 256, 0, stream>>>(x, w1, w2, counts, elist, ewt, out);
}
